// SSDLayer_52003464020562
// MI455X (gfx1250) — matrix operand order repaired and checked by reading, not compiled, not run on hardware
//
#include <hip/hip_runtime.h>

// Depthwise causal conv: out[b,l,h] = sum_{d<64} u[b,l-d,h] * c[h,d]
// B=16, L=4096, H=256, D=64.
// Toeplitz-matmul mapping, 4 output sub-tiles per staged window:
//   window: j in [l0-80, l0+64)  (144 rows) covers 64 outputs (4 x 16)
//   A_c[m,k32] = c[h, m+80-32c-k32]   -- shared by all sub-tiles (relative band)
//   sub-tile s reads its B slice at window offset 16s+32c
//   12 x v_wmma_f32_16x16x32_f16, 4 independent accumulator chains.
// CDNA5 data movement: filter staging via global_load_async_to_lds_b128,
// result drain via global_store_async_from_lds_b128 th:TH_STORE_NT (ASYNCcnt).

#define LL 4096
#define HH 256
#define DD 64

typedef __attribute__((ext_vector_type(16))) _Float16 v16h;
typedef __attribute__((ext_vector_type(8)))  _Float16 v8h;
typedef __attribute__((ext_vector_type(8)))  float    v8f;

#define NCH 8                 // channels (waves) per block
#define NTHREADS 256
#define WIN 144               // window rows: [l0-80, l0+64)
#define SUBT 4                // 4 sub-tiles of 16 l-positions = 64 outputs / window
#define B_STRIDE 152          // halfs per batch row (144 + 8 pad, 16B-aligned rows)
#define H_STRIDE 2440         // halfs per channel row (16*152 + 8 pad; bank stride 4)
#define SMEM_FLOATS 9760      // NCH*H_STRIDE halfs / 2 = 39040 B (reused per phase)
#define OUT_B_STRIDE 516      // f32 words per batch row in output staging (16B-aligned)
#define WINDOWS_PER_BLOCK 2

__global__ __launch_bounds__(NTHREADS)
void ssd_conv_wmma(const float* __restrict__ u,
                   const float* __restrict__ cflt,
                   float* __restrict__ out) {
    __shared__ __align__(16) float smem[SMEM_FLOATS];
    _Float16* uh = reinterpret_cast<_Float16*>(smem);
    // 32-bit LDS offset of smem (generic-pointer low bits = LDS offset on gfx1250)
    const unsigned smemBase = (unsigned)(size_t)(const void*)&smem[0];

    const int t      = threadIdx.x;
    const int wv     = t >> 5;            // wave id == channel offset (wave32)
    const int lane   = t & 31;
    const int laneHi = (lane >= 16) ? 1 : 0;
    const int nb     = lane & 15;         // N (batch) index; also M for A build

    const int h0     = blockIdx.x * NCH;
    const int lBlock = blockIdx.y * (16 * SUBT * WINDOWS_PER_BLOCK);

    // ---- Async-stage the block's 8 filters (512 f32) into LDS: 128 x b128.
    if (t < (NCH * DD) / 4) {
        const unsigned long long g =
            (unsigned long long)(const void*)(cflt + (size_t)h0 * DD + t * 4);
        const unsigned l = smemBase + (unsigned)t * 16u;
        asm volatile("global_load_async_to_lds_b128 %0, %1, off"
                     :: "v"(l), "v"(g) : "memory");
    }
    asm volatile("s_wait_asynccnt 0x0" ::: "memory");
    __syncthreads();

    // ---- Branchless Toeplitz A build from LDS: 3 chunks of f16 16x32.
    // A layout (16-bit A): elem i -> K = (i/8)*16 + laneHi*8 + (i%8), M = lane&15.
    v16h a[3];
#pragma unroll
    for (int c = 0; c < 3; ++c) {
#pragma unroll
        for (int i = 0; i < 16; ++i) {
            const int K = c * 32 + (i >> 3) * 16 + laneHi * 8 + (i & 7);
            const int d = nb + 80 - K;                      // tap index, in [-15, 95]
            const float v = smem[wv * DD + (d & (DD - 1))]; // clamped ds_load
            a[c][i] = (_Float16)(((unsigned)d < (unsigned)DD) ? v : 0.0f);
        }
    }

    // Per-thread staging coords are loop-invariant (stride NTHREADS preserves them).
    const int s_hh  = t & 7;
    const int s_b   = (t >> 3) & 15;
    const int s_jj0 = t >> 7;             // 0 or 1; jj = s_jj0 + 2*step
    const float* uBase =
        u + (size_t)s_b * (LL * HH) + (size_t)s_hh + h0;

    for (int w = 0; w < WINDOWS_PER_BLOCK; ++w) {
        const int l0 = lBlock + w * (16 * SUBT);

        // drain previous window's async stores before reusing smem
        asm volatile("s_wait_asynccnt 0x0" ::: "memory");
        __syncthreads();

        // ---- Stage u window [l0-80, l0+64) x 16 b x 8 ch as f16 (branchless,
        // 8-wide load batches so global loads clause before the converts/stores).
#pragma unroll 1
        for (int batch = 0; batch < (WIN * 16 * NCH) / (NTHREADS * 8); ++batch) {
            float vbuf[8];
#pragma unroll
            for (int k = 0; k < 8; ++k) {
                const int jj = s_jj0 + 2 * (batch * 8 + k);
                const int j  = l0 - 80 + jj;
                const int jc = (j >= 0) ? j : 0;            // clamped address
                vbuf[k] = uBase[(size_t)jc * HH];
            }
#pragma unroll
            for (int k = 0; k < 8; ++k) {
                const int jj = s_jj0 + 2 * (batch * 8 + k);
                const int j  = l0 - 80 + jj;
                const float v = (j >= 0) ? vbuf[k] : 0.0f;  // v_cndmask, no branch
                uh[s_hh * H_STRIDE + s_b * B_STRIDE + jj] = (_Float16)v;
            }
        }
        __syncthreads();

        // ---- 4 sub-tiles x 3 K-chunks = 12 WMMAs, 4 independent acc chains.
        // B layout (16-bit B): elem i -> K = (i/8)*16 + laneHi*8 + (i%8), N = lane&15.
        v8f acc[SUBT];
#pragma unroll
        for (int s = 0; s < SUBT; ++s) acc[s] = v8f{};
#pragma unroll
        for (int c = 0; c < 3; ++c) {
#pragma unroll
            for (int s = 0; s < SUBT; ++s) {
                const int Kbase = 16 * s + 32 * c + 8 * laneHi;
                const v8h* p = reinterpret_cast<const v8h*>(
                    &uh[wv * H_STRIDE + nb * B_STRIDE + Kbase]); // 16B aligned
                const v8h blo = p[0];
                const v8h bhi = p[2];
                const v16h bmat = __builtin_shufflevector(
                    blo, bhi, 0, 1, 2, 3, 4, 5, 6, 7, 8, 9, 10, 11, 12, 13, 14, 15);
                acc[s] = __builtin_amdgcn_wmma_f32_16x16x32_f16(
                    false, a[c], false, bmat, (short)0, acc[s], false, false);
            }
        }

        __syncthreads();

        // ---- Stage D tiles (f32) to LDS: layout [b][m][h], rows 16B-aligned.
        // D layout: elem r -> M = r + 8*laneHi, N = lane&15 (= batch).
#pragma unroll
        for (int s = 0; s < SUBT; ++s) {
#pragma unroll
            for (int r = 0; r < 8; ++r) {
                const int m = 16 * s + r + laneHi * 8;
                smem[nb * OUT_B_STRIDE + m * NCH + wv] = acc[s][r];
            }
        }
        __syncthreads();

        // ---- Drain to global with async LDS->global b128 stores, NT policy
        // (output is streamed once; keep u resident in L2).
#pragma unroll
        for (int k = 0; k < (16 * 16 * SUBT * NCH) / (NTHREADS * 4); ++k) {
            const int q    = t + k * NTHREADS;
            const int part = q & 1;                 // which 16B half of the 32B row
            const int b    = (q >> 1) & 15;
            const int m    = q >> 5;                // 0..63
            const unsigned long long g = (unsigned long long)(const void*)
                (out + (size_t)b * (LL * HH) + (size_t)(l0 + m) * HH + h0 + part * 4);
            const unsigned l =
                smemBase + (unsigned)(b * OUT_B_STRIDE + m * NCH + part * 4) * 4u;
            asm volatile("global_store_async_from_lds_b128 %0, %1, off th:TH_STORE_NT"
                         :: "v"(g), "v"(l) : "memory");
        }
    }
    // implicit wait-idle at s_endpgm covers the final async stores
}

extern "C" void kernel_launch(void* const* d_in, const int* in_sizes, int n_in,
                              void* d_out, int out_size, void* d_ws, size_t ws_size,
                              hipStream_t stream) {
    const float* u = (const float*)d_in[0];   // [B, L, H] f32
    const float* c = (const float*)d_in[1];   // [C=1, H, D] f32
    float* out = (float*)d_out;               // [B, L, H] f32

    dim3 grid(HH / NCH, LL / (16 * SUBT * WINDOWS_PER_BLOCK));  // (32, 32)
    ssd_conv_wmma<<<grid, NTHREADS, 0, stream>>>(u, c, out);
}
